// PowerModel_84061099917518
// MI455X (gfx1250) — matrix operand order repaired and checked by reading, not compiled, not run on hardware
//
#include <hip/hip_runtime.h>
#include <hip/hip_bf16.h>
#include <math.h>

typedef __attribute__((ext_vector_type(16))) _Float16 v16h;
typedef __attribute__((ext_vector_type(8)))  _Float16 v8h;
typedef __attribute__((ext_vector_type(8)))  float    v8f;
typedef __attribute__((ext_vector_type(8)))  int      v8i;
typedef __attribute__((ext_vector_type(4)))  int      v4i;
typedef __attribute__((ext_vector_type(4)))  unsigned int v4ui;

#define NB      64
#define SEQ     192
#define DECL    36
#define CIN     21
#define MARKN   4
#define DM      512
#define DFF     2048
#define LBL     12
#define PRED    24

#if defined(__has_builtin)
#if __has_builtin(__builtin_amdgcn_tensor_load_to_lds) && __has_builtin(__builtin_amdgcn_s_wait_tensorcnt)
#define USE_TDM 1
#endif
#endif

// ---------------------------------------------------------------------------
// f32 -> f16 staging
// ---------------------------------------------------------------------------
__global__ void k_cvt_f16(const float* __restrict__ src, _Float16* __restrict__ dst, int n) {
  int i = blockIdx.x * blockDim.x + threadIdx.x;
  if (i < n) dst[i] = (_Float16)src[i];
}

// ---------------------------------------------------------------------------
// WMMA GEMM: C[m][n] = sum_k A[m][k]*W[n][k] + bias[n]   (A @ W^T)
// A: MxK f16 row-major, W: NxK f16 row-major.
// Block = 128 threads = 4 waves; block tile = 64(M) x 32(N).
// Per wave: 16(M) x 32(N) via two v_wmma_f32_16x16x32_f16 per 32-wide K step.
// B tile (32 rows x 32 K, f16) staged into LDS once per block per K step:
//   - TDM path: one wave issues TENSOR_LOAD_TO_LDS (2D descriptor), waits
//     s_wait_tensorcnt, then workgroup barrier.
//   - fallback: cooperative b128 vector loads.
// Requires: M % 64 == 0, N % 32 == 0, K % 32 == 0 (true for all call sites).
// ---------------------------------------------------------------------------
__global__ __launch_bounds__(128)
void k_wmma_gemm(const _Float16* __restrict__ A, const _Float16* __restrict__ W,
                 const float* __restrict__ bias, float* __restrict__ C,
                 int M, int N, int K) {
  (void)M;
  __shared__ _Float16 tileB[32][32];           // 2KB: rows = N cols, 32 K each

  const int lane = threadIdx.x & 31;
  const int wv   = threadIdx.x >> 5;           // 0..3
  const int half = lane >> 4;                  // 0: lanes 0-15, 1: lanes 16-31
  const int l16  = lane & 15;
  const int tn0  = blockIdx.x * 32;
  const int tm   = blockIdx.y * 64 + wv * 16;

  const _Float16* arow = A + (size_t)(tm + l16) * K;

  v8f acc0 = {}, acc1 = {};
  for (int k0 = 0; k0 < K; k0 += 32) {
#if defined(USE_TDM)
    if (wv == 0) {
      // Tensor DMA descriptor (CDNA5 ISA ch.8): 2D tile, 2-byte elements,
      // tile 32(K) x 32(N rows), tensor row stride = K elements.
      unsigned lds_base = (unsigned)(size_t)(&tileB[0][0]);
      unsigned long long ga =
          (unsigned long long)(size_t)(W + (size_t)tn0 * K + k0);
      v4ui g0;
      g0.x = 1u;                                            // count=1 (valid)
      g0.y = lds_base;                                      // lds_addr
      g0.z = (unsigned)(ga & 0xffffffffull);                // global_addr lo
      g0.w = (unsigned)((ga >> 32) & 0x01ffffffull) | (2u << 30); // hi | type=2
      unsigned Klo = (unsigned)(K & 0xffff), Khi = (unsigned)((K >> 16) & 0xffff);
      unsigned Nlo = (unsigned)(N & 0xffff), Nhi = (unsigned)((N >> 16) & 0xffff);
      v8i g1;
      g1[0] = (int)(1u << 16);               // data_size=1 (2B), mask=0
      g1[1] = (int)(Klo << 16);              // tensor_dim0[15:0]
      g1[2] = (int)(Khi | (Nlo << 16));      // tensor_dim0[31:16] | tensor_dim1[15:0]
      g1[3] = (int)(Nhi | (32u << 16));      // tensor_dim1[31:16] | tile_dim0=32
      g1[4] = (int)32;                       // tile_dim1=32, tile_dim2=0
      g1[5] = (int)K;                        // tensor_dim0_stride[31:0]
      g1[6] = 0;                             // stride0 hi | stride1 lo
      g1[7] = 0;
      v4i gz4 = {0, 0, 0, 0};
      v8i gz8 = {0, 0, 0, 0, 0, 0, 0, 0};
      __builtin_amdgcn_tensor_load_to_lds(g0, g1, gz4, gz4, gz8, 0);
      __builtin_amdgcn_s_wait_tensorcnt(0);
    }
    __syncthreads();
#else
    {
      int idx = threadIdx.x * 8;             // 128*8 = 1024 = 32x32 halves
      int n = idx >> 5, kk = idx & 31;
      *(v8h*)&tileB[n][kk] =
          *(const v8h*)&W[(size_t)(tn0 + n) * K + k0 + kk];
    }
    __syncthreads();
#endif
    __builtin_prefetch(arow + k0 + 128, 0, 0);

    // A 16x32 f16 lane layout: half 0 -> K{0..7,16..23}, half 1 -> K{8..15,24..31}
    v8h alo = *(const v8h*)&arow[k0 + half * 8];
    v8h ahi = *(const v8h*)&arow[k0 + 16 + half * 8];
    v16h a;
#pragma unroll
    for (int i = 0; i < 8; ++i) { a[i] = alo[i]; a[i + 8] = ahi[i]; }

    // B 32x16 f16 lane layout: half 0 -> K{0..7,16..23}, half 1 -> K{8..15,24..31}
    v8h b0lo = *(const v8h*)&tileB[l16][half * 8];
    v8h b0hi = *(const v8h*)&tileB[l16][16 + half * 8];
    v8h b1lo = *(const v8h*)&tileB[16 + l16][half * 8];
    v8h b1hi = *(const v8h*)&tileB[16 + l16][16 + half * 8];
    v16h b0, b1;
#pragma unroll
    for (int i = 0; i < 8; ++i) {
      b0[i] = b0lo[i]; b0[i + 8] = b0hi[i];
      b1[i] = b1lo[i]; b1[i + 8] = b1hi[i];
    }
    acc0 = __builtin_amdgcn_wmma_f32_16x16x32_f16(false, a, false, b0,
                                                  (short)0, acc0, false, false);
    acc1 = __builtin_amdgcn_wmma_f32_16x16x32_f16(false, a, false, b1,
                                                  (short)0, acc1, false, false);
    __syncthreads();
  }

  float bb0 = bias ? bias[tn0 + l16] : 0.0f;
  float bb1 = bias ? bias[tn0 + 16 + l16] : 0.0f;
#pragma unroll
  for (int i = 0; i < 8; ++i) {
    int r = tm + half * 8 + i;     // D layout: VGPR i -> row i (+8 upper half)
    C[(size_t)r * N + tn0 + l16]      = acc0[i] + bb0;
    C[(size_t)r * N + tn0 + 16 + l16] = acc1[i] + bb1;
  }
}

// ---------------------------------------------------------------------------
// Series decomposition: moving average (k=11, edge-replicate) -> trend,
// residual -> seasonal. Either output may be null.
// ---------------------------------------------------------------------------
__global__ void k_decomp(const float* __restrict__ x, float* __restrict__ seas,
                         float* __restrict__ trend, int B, int L, int C) {
  int idx = blockIdx.x * blockDim.x + threadIdx.x;
  int total = B * L * C;
  if (idx >= total) return;
  int c = idx % C; int t = (idx / C) % L; int b = idx / (C * L);
  const float* base = x + (size_t)b * L * C + c;
  float s = 0.f;
#pragma unroll
  for (int j = -5; j <= 5; ++j) {
    int ti = t + j; ti = ti < 0 ? 0 : (ti >= L ? L - 1 : ti);
    s += base[(size_t)ti * C];
  }
  float m = s * (1.0f / 11.0f);
  if (seas)  seas[idx]  = x[idx] - m;
  if (trend) trend[idx] = m;
}

// ---------------------------------------------------------------------------
// Embedding: circular conv1d (width 3) over Cin channels + mark @ time_w^T
// ---------------------------------------------------------------------------
__global__ __launch_bounds__(512)
void k_embed(const float* __restrict__ x, const float* __restrict__ mark,
             const float* __restrict__ convw, const float* __restrict__ timew,
             float* __restrict__ out, int L, int Cin) {
  int bl = blockIdx.x; int b = bl / L; int l = bl % L; int f = threadIdx.x;
  float s = 0.f;
  for (int j = 0; j < 3; ++j) {
    int li = l + j - 1; li = (li + L) % L;
    const float* xr = x + ((size_t)b * L + li) * Cin;
    const float* wr = convw + (size_t)f * Cin * 3 + j;  // convw[f][c][j]
    for (int c = 0; c < Cin; ++c) s += xr[c] * wr[(size_t)c * 3];
  }
  const float* mr = mark + ((size_t)b * L + l) * MARKN;
#pragma unroll
  for (int m = 0; m < MARKN; ++m) s += mr[m] * timew[(size_t)f * MARKN + m];
  out[((size_t)b * L + l) * DM + f] = s;
}

// ---------------------------------------------------------------------------
// Trend projection: circular conv1d (width 3), DM channels -> CIN channels
// ---------------------------------------------------------------------------
__global__ __launch_bounds__(32)
void k_trendconv(const float* __restrict__ x, const float* __restrict__ w,
                 float* __restrict__ out, int L) {
  int bl = blockIdx.x; int b = bl / L; int l = bl % L; int c = threadIdx.x;
  if (c >= CIN) return;
  float s = 0.f;
  for (int j = 0; j < 3; ++j) {
    int li = l + j - 1; li = (li + L) % L;
    const float* xr = x + ((size_t)b * L + li) * DM;
    const float* wr = w + (size_t)c * DM * 3 + j;       // w[c][d][j]
    for (int d = 0; d < DM; ++d) s += xr[d] * wr[(size_t)d * 3];
  }
  out[((size_t)b * L + l) * CIN + c] = s;
}

// ---------------------------------------------------------------------------
// Circular autocorrelation: corr[b][tau] = mean_d sum_t q[t,d]*k[(t-tau)%L,d]
// ---------------------------------------------------------------------------
__global__ __launch_bounds__(256)
void k_corr(const float* __restrict__ q, const float* __restrict__ k,
            float* __restrict__ corr, int L) {
  int b = blockIdx.x / L; int tau = blockIdx.x % L;
  float s = 0.f;
  int total = L * DM;
  for (int i = threadIdx.x; i < total; i += 256) {
    int t = i / DM; int d = i - t * DM;
    int ts = t - tau; if (ts < 0) ts += L;
    s += q[((size_t)b * L + t) * DM + d] * k[((size_t)b * L + ts) * DM + d];
  }
  __shared__ float red[256];
  red[threadIdx.x] = s;
  __syncthreads();
  for (int st = 128; st > 0; st >>= 1) {
    if (threadIdx.x < st) red[threadIdx.x] += red[threadIdx.x + st];
    __syncthreads();
  }
  if (threadIdx.x == 0) corr[blockIdx.x] = red[0] * (1.0f / (float)DM);
}

// ---------------------------------------------------------------------------
// Per-batch serial top-k (k<=5) + softmax over selected correlations
// ---------------------------------------------------------------------------
__global__ void k_topk(const float* __restrict__ corr, float* __restrict__ wgt,
                       int* __restrict__ delay, int L, int topk) {
  if (threadIdx.x != 0) return;
  int b = blockIdx.x;
  float vals[6]; int idxs[6];
  for (int i = 0; i < topk; ++i) { vals[i] = -3.0e38f; idxs[i] = 0; }
  for (int t = 0; t < L; ++t) {
    float v = corr[(size_t)b * L + t];
    if (v > vals[topk - 1]) {
      int p = topk - 1;
      while (p > 0 && vals[p - 1] < v) {
        vals[p] = vals[p - 1]; idxs[p] = idxs[p - 1]; --p;
      }
      vals[p] = v; idxs[p] = t;
    }
  }
  float mx = vals[0], sum = 0.f, e[6];
  for (int i = 0; i < topk; ++i) { e[i] = __expf(vals[i] - mx); sum += e[i]; }
  for (int i = 0; i < topk; ++i) {
    wgt[b * topk + i] = e[i] / sum;
    delay[b * topk + i] = idxs[i];
  }
}

// out[b,t,d] = sum_k wgt[b,k] * v[b,(t+delay[b,k])%L,d]
__global__ void k_agg(const float* __restrict__ v, const float* __restrict__ wgt,
                      const int* __restrict__ delay, float* __restrict__ out,
                      int L, int topk) {
  int idx = blockIdx.x * blockDim.x + threadIdx.x;
  int total = NB * L * DM;
  if (idx >= total) return;
  int d = idx % DM; int t = (idx / DM) % L; int b = idx / (DM * L);
  float s = 0.f;
  for (int kk = 0; kk < topk; ++kk) {
    int dt = t + delay[b * topk + kk]; if (dt >= L) dt -= L;
    s += wgt[b * topk + kk] * v[((size_t)b * L + dt) * DM + d];
  }
  out[idx] = s;
}

// ---------------------------------------------------------------------------
// Elementwise helpers
// ---------------------------------------------------------------------------
__global__ void k_add(const float* __restrict__ a, const float* __restrict__ b,
                      float* __restrict__ c, int n) {
  int i = blockIdx.x * blockDim.x + threadIdx.x;
  if (i < n) c[i] = a[i] + b[i];
}
__global__ void k_add3(const float* __restrict__ a, const float* __restrict__ b,
                       const float* __restrict__ c, float* __restrict__ d, int n) {
  int i = blockIdx.x * blockDim.x + threadIdx.x;
  if (i < n) d[i] = a[i] + b[i] + c[i];
}
__global__ void k_addinp(float* __restrict__ a, const float* __restrict__ b, int n) {
  int i = blockIdx.x * blockDim.x + threadIdx.x;
  if (i < n) a[i] += b[i];
}
__global__ void k_gelu(float* __restrict__ x, int n) {
  int i = blockIdx.x * blockDim.x + threadIdx.x;
  if (i < n) {
    float v = x[i];
    x[i] = 0.5f * v * (1.0f + erff(v * 0.70710678118654752f));
  }
}

// LayerNorm over last dim (DM), one block per row
__global__ __launch_bounds__(256)
void k_layernorm(const float* __restrict__ x, const float* __restrict__ g,
                 const float* __restrict__ bta, float* __restrict__ y) {
  const float* xr = x + (size_t)blockIdx.x * DM;
  float* yr = y + (size_t)blockIdx.x * DM;
  __shared__ float red[256];
  float s = 0.f;
  for (int d = threadIdx.x; d < DM; d += 256) s += xr[d];
  red[threadIdx.x] = s; __syncthreads();
  for (int st = 128; st > 0; st >>= 1) {
    if (threadIdx.x < st) red[threadIdx.x] += red[threadIdx.x + st];
    __syncthreads();
  }
  float mu = red[0] * (1.0f / DM);
  __syncthreads();
  float v = 0.f;
  for (int d = threadIdx.x; d < DM; d += 256) { float t = xr[d] - mu; v += t * t; }
  red[threadIdx.x] = v; __syncthreads();
  for (int st = 128; st > 0; st >>= 1) {
    if (threadIdx.x < st) red[threadIdx.x] += red[threadIdx.x + st];
    __syncthreads();
  }
  float inv = rsqrtf(red[0] * (1.0f / DM) + 1e-5f);
  __syncthreads();
  for (int d = threadIdx.x; d < DM; d += 256)
    yr[d] = (xr[d] - mu) * inv * g[d] + bta[d];
}

// subtract mean over seq dim (axis 1); one thread per (b,d)
__global__ void k_sub_seqmean(float* __restrict__ y, int L) {
  int idx = blockIdx.x * blockDim.x + threadIdx.x;
  if (idx >= NB * DM) return;
  int b = idx / DM; int d = idx % DM;
  float* base = y + (size_t)b * L * DM + d;
  float s = 0.f;
  for (int t = 0; t < L; ++t) s += base[(size_t)t * DM];
  s *= (1.0f / (float)L);
  for (int t = 0; t < L; ++t) base[(size_t)t * DM] -= s;
}

// Small GEMM (A @ W^T + bias), one thread per output; act: 0=none, 1=leaky
__global__ void k_small_gemm(const float* __restrict__ A, const float* __restrict__ W,
                             const float* __restrict__ bias, float* __restrict__ C,
                             int M, int N, int K, int act) {
  int idx = blockIdx.x * blockDim.x + threadIdx.x;
  if (idx >= M * N) return;
  int m = idx / N, n = idx % N;
  const float* ar = A + (size_t)m * K;
  const float* wr = W + (size_t)n * K;
  float s = bias ? bias[n] : 0.f;
  for (int k = 0; k < K; ++k) s += ar[k] * wr[k];
  if (act == 1) s = (s >= 0.f) ? s : 0.01f * s;
  C[idx] = s;
}

__global__ void k_mean_t(const float* __restrict__ x, float* __restrict__ mean) {
  int idx = blockIdx.x * blockDim.x + threadIdx.x;
  if (idx >= NB * CIN) return;
  int b = idx / CIN, c = idx % CIN;
  float s = 0.f;
  for (int t = 0; t < SEQ; ++t) s += x[((size_t)b * SEQ + t) * CIN + c];
  mean[idx] = s * (1.0f / SEQ);
}

__global__ void k_build_inits(const float* __restrict__ seas_raw,
                              const float* __restrict__ trend_raw,
                              const float* __restrict__ mean,
                              float* __restrict__ seas_init,
                              float* __restrict__ trend_init) {
  int idx = blockIdx.x * blockDim.x + threadIdx.x;
  if (idx >= NB * DECL * CIN) return;
  int c = idx % CIN; int t = (idx / CIN) % DECL; int b = idx / (CIN * DECL);
  if (t < LBL) {
    int src = ((size_t)b * SEQ + (SEQ - LBL + t)) * CIN + c;
    seas_init[idx]  = seas_raw[src];
    trend_init[idx] = trend_raw[src];
  } else {
    seas_init[idx]  = 0.f;
    trend_init[idx] = mean[b * CIN + c];
  }
}

// gather first DECL rows per batch from a [NB,SEQ,DM] buffer
__global__ void k_take_rows(const float* __restrict__ src, float* __restrict__ dst) {
  int idx = blockIdx.x * blockDim.x + threadIdx.x;
  if (idx >= NB * DECL * DM) return;
  int d = idx % DM; int t = (idx / DM) % DECL; int b = idx / (DM * DECL);
  dst[idx] = src[((size_t)b * SEQ + t) * DM + d];
}

__global__ void k_slice_out(const float* __restrict__ h, float* __restrict__ out) {
  int idx = blockIdx.x * blockDim.x + threadIdx.x;
  if (idx >= NB * PRED) return;
  int b = idx / PRED, t = idx % PRED;
  out[idx] = h[(size_t)b * DECL + LBL + t];
}

// ---------------------------------------------------------------------------
// Host orchestration
// ---------------------------------------------------------------------------
struct AttnP { const float *wq,*bq,*wk,*bk,*wv,*bv,*wo,*bo; };

extern "C" void kernel_launch(void* const* d_in, const int* in_sizes, int n_in,
                              void* d_out, int out_size, void* d_ws, size_t ws_size,
                              hipStream_t stream) {
  (void)in_sizes; (void)n_in; (void)out_size; (void)ws_size;

  const float* x_enc      = (const float*)d_in[0];
  const float* x_mark_enc = (const float*)d_in[1];
  // d_in[2] = x_dec: numerically unused by the reference (shape only)
  const float* x_mark_dec = (const float*)d_in[3];

  // params flattened in reference-dict insertion order
  int pi = 4;
  auto P = [&](void) { return (const float*)d_in[pi++]; };
  const float* enc_conv_w = P(); const float* enc_time_w = P();
  const float* dec_conv_w = P(); const float* dec_time_w = P();
  auto getAttn = [&](void) {
    AttnP a;
    a.wq = P(); a.bq = P(); a.wk = P(); a.bk = P();
    a.wv = P(); a.bv = P(); a.wo = P(); a.bo = P();
    return a;
  };
  AttnP enc_attn[2]; const float* enc_c1[2]; const float* enc_c2[2];
  for (int l = 0; l < 2; ++l) { enc_attn[l] = getAttn(); enc_c1[l] = P(); enc_c2[l] = P(); }
  AttnP dec_self = getAttn(), dec_cross = getAttn();
  const float* dec_c1 = P(); const float* dec_c2 = P(); const float* trend_w = P();
  const float* enc_g = P(); const float* enc_b = P();
  const float* dec_g = P(); const float* dec_b = P();
  const float* proj_w = P(); const float* proj_b = P();
  const float* lin1_w = P(); const float* lin1_b = P();
  const float* lin2_w = P(); const float* lin2_b = P();

  // workspace carve-up (floats, each region padded to 32B alignment)
  size_t off = 0;
  auto alloc = [&](size_t n) {
    n = (n + 7) & ~(size_t)7;
    float* p = (float*)d_ws + off; off += n; return p;
  };
  const size_t ENC = (size_t)NB * SEQ * DM;
  const size_t FFN = (size_t)NB * SEQ * DFF;
  const size_t DEC = (size_t)NB * DECL * DM;
  float* enc_x  = alloc(ENC);
  float* bq     = alloc(ENC);
  float* bk     = alloc(ENC);
  float* bv     = alloc(ENC);
  float* battn  = alloc(ENC);
  float* btmp   = alloc(ENC);
  float* bffn   = alloc(FFN);
  _Float16* a16 = (_Float16*)alloc(FFN / 2);          // f16 staging: activations
  _Float16* w16 = (_Float16*)alloc((size_t)DFF * DM / 2); // f16 staging: weights
  float* dec_x  = alloc(DEC);
  float* dt1    = alloc(DEC);
  float* dt2    = alloc(DEC);
  float* dt3    = alloc(DEC);
  float* dtmp   = alloc(DEC);
  float* enc36  = alloc(DEC);
  float* corr   = alloc((size_t)NB * SEQ);
  float* wgt    = alloc((size_t)NB * 8);
  int*   delay  = (int*)alloc((size_t)NB * 8);
  float* meanbc = alloc((size_t)NB * CIN);
  float* sraw   = alloc((size_t)NB * SEQ * CIN);
  float* traw   = alloc((size_t)NB * SEQ * CIN);
  float* sinit  = alloc((size_t)NB * DECL * CIN);
  float* trend  = alloc((size_t)NB * DECL * CIN);
  float* rt     = alloc((size_t)NB * DECL * CIN);
  float* osum   = alloc((size_t)NB * DECL * CIN);
  float* h1     = alloc((size_t)NB * DECL * 256);
  float* h2     = alloc((size_t)NB * DECL);

  // WMMA GEMM wrapper: stage A and W to f16, then tiled WMMA kernel.
  auto gemm = [&](const float* A, const float* Wm, const float* bias, float* C,
                  int M, int N, int K) {
    int na = M * K, nw = N * K;
    k_cvt_f16<<<(na + 255) / 256, 256, 0, stream>>>(A, a16, na);
    k_cvt_f16<<<(nw + 255) / 256, 256, 0, stream>>>(Wm, w16, nw);
    dim3 g(N / 32, M / 64);
    k_wmma_gemm<<<g, 128, 0, stream>>>(a16, w16, bias, C, M, N, K);
  };
  auto attention = [&](const float* qx, const float* kvx, int L, int topk,
                       const AttnP& a, float* out_buf) {
    int M = NB * L;
    gemm(qx,  a.wq, a.bq, bq, M, DM, DM);
    gemm(kvx, a.wk, a.bk, bk, M, DM, DM);
    gemm(kvx, a.wv, a.bv, bv, M, DM, DM);
    k_corr<<<NB * L, 256, 0, stream>>>(bq, bk, corr, L);
    k_topk<<<NB, 32, 0, stream>>>(corr, wgt, delay, L, topk);
    int tot = NB * L * DM;
    k_agg<<<(tot + 255) / 256, 256, 0, stream>>>(bv, wgt, delay, btmp, L, topk);
    gemm(btmp, a.wo, a.bo, out_buf, M, DM, DM);
  };

  // --- decomposition of raw encoder input + decoder inits ---
  {
    int n = NB * SEQ * CIN;
    k_decomp<<<(n + 255) / 256, 256, 0, stream>>>(x_enc, sraw, traw, NB, SEQ, CIN);
    k_mean_t<<<(NB * CIN + 255) / 256, 256, 0, stream>>>(x_enc, meanbc);
    int m = NB * DECL * CIN;
    k_build_inits<<<(m + 255) / 256, 256, 0, stream>>>(sraw, traw, meanbc, sinit, trend);
  }

  // --- encoder ---
  k_embed<<<NB * SEQ, 512, 0, stream>>>(x_enc, x_mark_enc, enc_conv_w, enc_time_w,
                                        enc_x, SEQ, CIN);
  const int nE = NB * SEQ * DM;
  for (int l = 0; l < 2; ++l) {
    attention(enc_x, enc_x, SEQ, 5, enc_attn[l], battn);
    k_add<<<(nE + 255) / 256, 256, 0, stream>>>(enc_x, battn, btmp, nE);
    k_decomp<<<(nE + 255) / 256, 256, 0, stream>>>(btmp, bq, nullptr, NB, SEQ, DM);
    gemm(bq, enc_c1[l], nullptr, bffn, NB * SEQ, DFF, DM);
    int nf = NB * SEQ * DFF;
    k_gelu<<<(nf + 255) / 256, 256, 0, stream>>>(bffn, nf);
    gemm(bffn, enc_c2[l], nullptr, bk, NB * SEQ, DM, DFF);
    k_add<<<(nE + 255) / 256, 256, 0, stream>>>(bq, bk, bv, nE);
    k_decomp<<<(nE + 255) / 256, 256, 0, stream>>>(bv, enc_x, nullptr, NB, SEQ, DM);
  }
  k_layernorm<<<NB * SEQ, 256, 0, stream>>>(enc_x, enc_g, enc_b, btmp);
  k_sub_seqmean<<<(NB * DM + 255) / 256, 256, 0, stream>>>(btmp, SEQ);
  k_take_rows<<<(NB * DECL * DM + 255) / 256, 256, 0, stream>>>(btmp, enc36);

  // --- decoder ---
  k_embed<<<NB * DECL, 512, 0, stream>>>(sinit, x_mark_dec, dec_conv_w, dec_time_w,
                                         dec_x, DECL, CIN);
  const int nD = NB * DECL * DM;
  {
    attention(dec_x, dec_x, DECL, 3, dec_self, battn);
    k_add<<<(nD + 255) / 256, 256, 0, stream>>>(dec_x, battn, dtmp, nD);
    k_decomp<<<(nD + 255) / 256, 256, 0, stream>>>(dtmp, dec_x, dt1, NB, DECL, DM);

    attention(dec_x, enc36, DECL, 3, dec_cross, battn);
    k_add<<<(nD + 255) / 256, 256, 0, stream>>>(dec_x, battn, dtmp, nD);
    k_decomp<<<(nD + 255) / 256, 256, 0, stream>>>(dtmp, dec_x, dt2, NB, DECL, DM);

    gemm(dec_x, dec_c1, nullptr, bffn, NB * DECL, DFF, DM);
    int nf = NB * DECL * DFF;
    k_gelu<<<(nf + 255) / 256, 256, 0, stream>>>(bffn, nf);
    gemm(bffn, dec_c2, nullptr, bk, NB * DECL, DM, DFF);
    k_add<<<(nD + 255) / 256, 256, 0, stream>>>(dec_x, bk, dtmp, nD);
    k_decomp<<<(nD + 255) / 256, 256, 0, stream>>>(dtmp, dec_x, dt3, NB, DECL, DM);

    k_add3<<<(nD + 255) / 256, 256, 0, stream>>>(dt1, dt2, dt3, dtmp, nD);
    k_trendconv<<<NB * DECL, 32, 0, stream>>>(dtmp, trend_w, rt, DECL);
    int nT = NB * DECL * CIN;
    k_addinp<<<(nT + 255) / 256, 256, 0, stream>>>(trend, rt, nT);
  }
  k_layernorm<<<NB * DECL, 256, 0, stream>>>(dec_x, dec_g, dec_b, dtmp);
  k_sub_seqmean<<<(NB * DM + 255) / 256, 256, 0, stream>>>(dtmp, DECL);

  // --- heads ---
  int nT = NB * DECL * CIN;
  k_small_gemm<<<(nT + 255) / 256, 256, 0, stream>>>(dtmp, proj_w, proj_b, rt,
                                                     NB * DECL, CIN, DM, 0);
  k_add<<<(nT + 255) / 256, 256, 0, stream>>>(trend, rt, osum, nT);
  int n1 = NB * DECL * 256;
  k_small_gemm<<<(n1 + 255) / 256, 256, 0, stream>>>(osum, lin1_w, lin1_b, h1,
                                                     NB * DECL, 256, CIN, 1);
  int n2 = NB * DECL;
  k_small_gemm<<<(n2 + 255) / 256, 256, 0, stream>>>(h1, lin2_w, lin2_b, h2,
                                                     NB * DECL, 1, 256, 0);
  k_slice_out<<<(NB * PRED + 255) / 256, 256, 0, stream>>>(h2, (float*)d_out);
}
